// GCNModel_24867860644186
// MI455X (gfx1250) — matrix operand order repaired and checked by reading, not compiled, not run on hardware
//
#include <hip/hip_runtime.h>
#include <hip/hip_bf16.h>

// GAT layer + edge scorer for MI455X (gfx1250, wave32).
// Heavy GEMM runs on v_wmma_f32_16x16x32_f16 with the A tile DMA'd into LDS
// by the Tensor Data Mover (tensor_load_to_lds + s_wait_tensorcnt); edge
// phases are L2-resident gather/atomic kernels (xp/h tables fit in 192MB L2).

typedef __attribute__((ext_vector_type(16))) _Float16 v16h;
typedef __attribute__((ext_vector_type(8)))  _Float16 v8h;
typedef __attribute__((ext_vector_type(8)))  float    v8f;
typedef __attribute__((ext_vector_type(4)))  unsigned int v4u;
typedef __attribute__((ext_vector_type(8)))  int      v8i;
typedef __attribute__((ext_vector_type(4)))  int      v4i;

#define HEADS 4
#define HID   64
#define DIN   256
#define DOUT  256   // HEADS*HID

#if __has_builtin(__builtin_amdgcn_tensor_load_to_lds) && \
    __has_builtin(__builtin_amdgcn_s_wait_tensorcnt)
#define USE_TDM 1
#else
#define USE_TDM 0
#endif

// ---------------------------------------------------------------- helpers
__device__ __forceinline__ void atomicMaxF(float* addr, float v) {
  // monotone bit trick: ints order positives, flipped uints order negatives.
  if (!(v < 0.0f)) atomicMax((int*)addr, __float_as_int(v));
  else             atomicMin((unsigned int*)addr, __float_as_uint(v));
}

// ---------------------------------------------------------------- converts
__global__ void cvt_x_f16(const float* __restrict__ x, _Float16* __restrict__ xh, int total) {
  int i = blockIdx.x * blockDim.x + threadIdx.x;
  if (i < total) xh[i] = (_Float16)x[i];
}

// wt[n*256+k] = W[k*256+n]  (transpose so B fragments are contiguous in K)
__global__ void cvt_w_t_f16(const float* __restrict__ W, _Float16* __restrict__ wt) {
  int i = blockIdx.x * blockDim.x + threadIdx.x;   // 65536 elems
  if (i >= DIN * DOUT) return;
  int n = i >> 8, k = i & 255;
  wt[n * DIN + k] = (_Float16)W[k * DOUT + n];
}

// ---------------------------------------------------------------- WMMA GEMM
// xp[N,256] = xh[N,256] @ W[256,256]   (B fed from transposed wt[256,256])
// 1 block = 4 waves; block tile 16(M) x 256(N); A tile staged in LDS by TDM.
__global__ __launch_bounds__(128) void gat_gemm_xw(const _Float16* __restrict__ xh,
                                                   const _Float16* __restrict__ wt,
                                                   float* __restrict__ xp, int n) {
  __shared__ _Float16 As[16 * DIN];                 // 8 KB, LDS offset 0
  const int tid  = threadIdx.x;
  const int wave = tid >> 5;
  const int lane = tid & 31;
  const int m0   = blockIdx.x * 16;

#if USE_TDM
  if (tid < 32) {
    // Tensor DMA descriptor (CDNA5 ISA): 2-D tile,
    // 16 rows x 512B (64 x 8-byte elements), contiguous, into LDS offset 0.
    unsigned long long ga = (unsigned long long)(const void*)(xh + (size_t)m0 * DIN);
    v4u g0;
    g0[0] = 1u;                                       // count=1 (valid user D#)
    g0[1] = 0u;                                       // lds_addr = 0 (As)
    g0[2] = (unsigned)(ga & 0xffffffffull);           // global_addr[31:0]
    g0[3] = (unsigned)((ga >> 32) & 0x01ffffffull)    // global_addr[56:32]
            | (2u << 30);                             // type = 2 ("image")
    v8i g1 = (v8i)0;
    g1[0] = (3 << 16);          // workgroup_mask=0 (no cluster), data_size=3 (8B)
    g1[1] = (64 << 16);         // atomic_barrier_addr=0; tensor_dim0 lo16 = 64
    g1[2] = (16 << 16);         // tensor_dim0 hi16 = 0; tensor_dim1 lo16 = 16
    g1[3] = (64 << 16);         // tensor_dim1 hi16 = 0; tile_dim0 = 64
    g1[4] = 16;                 // tile_dim1 = 16 rows; tile_dim2 = 0
    g1[5] = 64;                 // tensor_dim0_stride = 64 (8B units, contiguous)
    g1[6] = 0;
    g1[7] = 0;
    v4i gz4 = (v4i)0;           // 2-D tensor: groups 2/3 unused
    v8i gz8 = (v8i)0;           // trailing group (6-arg toolchain form)
    __builtin_amdgcn_tensor_load_to_lds(g0, g1, gz4, gz4, gz8, 0);
    __builtin_amdgcn_s_wait_tensorcnt(0);             // TENSORcnt -> 0 (wave 0)
  }
  {
    // The TDM write to LDS is invisible to alias analysis (it goes through
    // the descriptor, not the As pointer). Escape As through an asm with a
    // memory clobber so the compiler treats the tile as written and keeps
    // the ds_load_b128 fragment reads alive.
    void* ap = (void*)As;
    asm volatile("" : "+v"(ap) :: "memory");
  }
  __syncthreads();
#else
  {   // fallback: cooperative 8KB copy, 512 uint4, 128 threads x 4
    const uint4* src = (const uint4*)(xh + (size_t)m0 * DIN);
    uint4* dst = (uint4*)As;
#pragma unroll
    for (int i = 0; i < 4; ++i) dst[tid + i * 128] = src[tid + i * 128];
  }
  __syncthreads();
#endif

  const int half = lane >> 4;        // 0..1
  const int ml   = lane & 15;        // row (A) / col (B) within tile
  const int nbase = wave * 64;

  v8f acc[4] = {v8f{}, v8f{}, v8f{}, v8f{}};

  for (int k0 = 0; k0 < DIN; k0 += 32) {
    // A fragment (16-bit 16x32 layout): elems 0..7 = K k0+8*half..+7,
    // elems 8..15 = K k0+16+8*half..+7, row = ml.  Two ds b128 reads.
    union { v16h v; v8h h[2]; } a;
    const _Float16* ap = As + ml * DIN + k0 + 8 * half;
    a.h[0] = *(const v8h*)(ap);
    a.h[1] = *(const v8h*)(ap + 16);

#pragma unroll
    for (int t = 0; t < 4; ++t) {
      // B fragment (16-bit 32x16 layout): col = ml (+n0), elems 0..7 =
      // K k0+8*half..+7, elems 8..15 = K k0+16+8*half..+7 in wt (transposed W). Two b128 loads.
      union { v16h v; v8h h[2]; } b;
      const _Float16* bp = wt + (size_t)(nbase + t * 16 + ml) * DIN + k0 + 8 * half;
      b.h[0] = *(const v8h*)(bp);
      b.h[1] = *(const v8h*)(bp + 16);
      acc[t] = __builtin_amdgcn_wmma_f32_16x16x32_f16(
                 false, a.v, false, b.v, (short)0, acc[t], false, false);
    }
  }

  // D layout: VGPR r, lanes 0-15 -> (M=r, N=lane); lanes 16-31 -> (M=r+8, N=lane-16)
#pragma unroll
  for (int t = 0; t < 4; ++t) {
    const int ncol = nbase + t * 16 + ml;
#pragma unroll
    for (int r = 0; r < 8; ++r) {
      int m = m0 + r + 8 * half;
      if (m < n) xp[(size_t)m * DOUT + ncol] = acc[t][r];
    }
  }
}

// ---------------------------------------------------------------- attn coeffs
__global__ void gat_attn_coeff(const float* __restrict__ xp,
                               const float* __restrict__ att_src,
                               const float* __restrict__ att_dst,
                               float* __restrict__ asrc, float* __restrict__ adst, int n) {
  int idx = blockIdx.x * blockDim.x + threadIdx.x;
  if (idx >= n * HEADS) return;
  int node = idx >> 2, h = idx & 3;
  const float* row = xp + (size_t)node * DOUT + h * HID;
  const float* ws_ = att_src + h * HID;
  const float* wd_ = att_dst + h * HID;
  float s = 0.f, t = 0.f;
#pragma unroll 8
  for (int c = 0; c < HID; ++c) { float v = row[c]; s += v * ws_[c]; t += v * wd_[c]; }
  asrc[idx] = s; adst[idx] = t;
}

// ---------------------------------------------------------------- init
__global__ void gat_init(float* __restrict__ agg, float* __restrict__ amax,
                         float* __restrict__ den, int n) {
  int idx = blockIdx.x * blockDim.x + threadIdx.x;
  if (idx < n * DOUT) agg[idx] = 0.f;
  if (idx < n * HEADS) { amax[idx] = __int_as_float(0xff800000); den[idx] = 0.f; }
}

// ---------------------------------------------------------------- softmax pass 1
__global__ void gat_edge_alpha(const int* __restrict__ src, const int* __restrict__ dst,
                               const float* __restrict__ asrc, const float* __restrict__ adst,
                               float* __restrict__ alph, float* __restrict__ amax,
                               int E, int n) {
  int idx = blockIdx.x * blockDim.x + threadIdx.x;
  int etot = E + n;
  if (idx >= etot * HEADS) return;
  int e = idx >> 2, h = idx & 3;
  int s, d;
  if (e < E) { s = src[e]; d = dst[e]; } else { s = d = e - E; }
  float a = asrc[s * HEADS + h] + adst[d * HEADS + h];
  a = (a > 0.f) ? a : 0.2f * a;                 // leaky_relu(0.2)
  alph[idx] = a;
  atomicMaxF(&amax[d * HEADS + h], a);
}

// ---------------------------------------------------------------- softmax pass 2
__global__ void gat_edge_exp(const int* __restrict__ dst,
                             const float* __restrict__ amax,
                             float* __restrict__ alph, float* __restrict__ den,
                             int E, int n) {
  int idx = blockIdx.x * blockDim.x + threadIdx.x;
  int etot = E + n;
  if (idx >= etot * HEADS) return;
  int e = idx >> 2, h = idx & 3;
  int d = (e < E) ? dst[e] : (e - E);
  float ex = __expf(alph[idx] - amax[d * HEADS + h]);
  alph[idx] = ex;
  atomicAdd(&den[d * HEADS + h], ex);
}

// ---------------------------------------------------------------- aggregate (wave per edge)
__global__ __launch_bounds__(256) void gat_scatter(const int* __restrict__ src,
                                                   const int* __restrict__ dst,
                                                   const float* __restrict__ xp,
                                                   const float* __restrict__ alph,
                                                   const float* __restrict__ den,
                                                   float* __restrict__ agg, int E, int n) {
  int e = blockIdx.x * 8 + (threadIdx.x >> 5);
  int lane = threadIdx.x & 31;
  int etot = E + n;
  if (e >= etot) return;
  int s, d;
  if (e < E) { s = src[e]; d = dst[e]; } else { s = d = e - E; }
  int h = lane >> 3;                               // 8 lanes per head
  float coef = alph[e * HEADS + h] / (den[d * HEADS + h] + 1e-16f);
  int c = lane * 8;
  const float4* xv = (const float4*)(xp + (size_t)s * DOUT + c);
  float4 v0 = xv[0], v1 = xv[1];
  float* o = agg + (size_t)d * DOUT + c;
  atomicAdd(o + 0, v0.x * coef); atomicAdd(o + 1, v0.y * coef);
  atomicAdd(o + 2, v0.z * coef); atomicAdd(o + 3, v0.w * coef);
  atomicAdd(o + 4, v1.x * coef); atomicAdd(o + 5, v1.y * coef);
  atomicAdd(o + 6, v1.z * coef); atomicAdd(o + 7, v1.w * coef);
}

// ---------------------------------------------------------------- bias + leaky(0.01), in place
__global__ void gat_bias_act(float* __restrict__ agg, const float* __restrict__ bias, int n) {
  int idx = blockIdx.x * blockDim.x + threadIdx.x;
  if (idx >= n * DOUT) return;
  float v = agg[idx] + bias[idx & 255];
  agg[idx] = (v > 0.f) ? v : 0.01f * v;
}

// ---------------------------------------------------------------- edge scorer (wave per edge)
__global__ __launch_bounds__(256) void edge_score(const int* __restrict__ src,
                                                  const int* __restrict__ dst,
                                                  const float* __restrict__ h,
                                                  const float* __restrict__ fc1W,
                                                  const float* __restrict__ fc1b,
                                                  float* __restrict__ out, int E) {
  int e = blockIdx.x * 8 + (threadIdx.x >> 5);
  int lane = threadIdx.x & 31;
  if (e >= E) return;
  int s = src[e], d = dst[e];
  int c = lane * 8;
  const float4* hs = (const float4*)(h + (size_t)s * DOUT + c);
  const float4* hd = (const float4*)(h + (size_t)d * DOUT + c);
  float4 a0 = hs[0], a1 = hs[1], b0 = hd[0], b1 = hd[1];
  float p[8] = {a0.x * b0.x, a0.y * b0.y, a0.z * b0.z, a0.w * b0.w,
                a1.x * b1.x, a1.y * b1.y, a1.z * b1.z, a1.w * b1.w};
  float s0 = 0.f, s1 = 0.f, s2 = 0.f;
#pragma unroll
  for (int i = 0; i < 8; ++i) {
    float pr = p[i] * 0.5f;
    const float* wr = fc1W + (c + i) * 3;
    s0 += pr * wr[0]; s1 += pr * wr[1]; s2 += pr * wr[2];
  }
#pragma unroll
  for (int off = 16; off > 0; off >>= 1) {        // wave32 tree reduce
    s0 += __shfl_down(s0, off, 32);
    s1 += __shfl_down(s1, off, 32);
    s2 += __shfl_down(s2, off, 32);
  }
  if (lane == 0) {
    out[(size_t)e * 3 + 0] = s0 + fc1b[0];
    out[(size_t)e * 3 + 1] = s1 + fc1b[1];
    out[(size_t)e * 3 + 2] = s2 + fc1b[2];
  }
}

// ---------------------------------------------------------------- launch
static inline size_t alignup(size_t v) { return (v + 255) & ~size_t(255); }

extern "C" void kernel_launch(void* const* d_in, const int* in_sizes, int n_in,
                              void* d_out, int out_size, void* d_ws, size_t ws_size,
                              hipStream_t stream) {
  const float* x       = (const float*)d_in[0];
  const int*   eidx    = (const int*)  d_in[1];
  const float* W       = (const float*)d_in[2];
  const float* att_src = (const float*)d_in[3];
  const float* att_dst = (const float*)d_in[4];
  const float* bias    = (const float*)d_in[5];
  const float* fc1W    = (const float*)d_in[6];
  const float* fc1b    = (const float*)d_in[7];
  float* out = (float*)d_out;

  const int N = in_sizes[0] / DIN;       // 50000
  const int E = in_sizes[1] / 2;         // 1600000
  const int ETOT = E + N;
  const int* src = eidx;                 // edge_index[0]
  const int* dst = eidx + E;             // edge_index[1]

  // workspace carve-out
  char* p = (char*)d_ws;
  _Float16* xh = (_Float16*)p; p += alignup((size_t)N * DIN * 2);
  _Float16* wt = (_Float16*)p; p += alignup((size_t)DIN * DOUT * 2);
  float* xp    = (float*)p;    p += alignup((size_t)N * DOUT * 4);
  float* asrc  = (float*)p;    p += alignup((size_t)N * HEADS * 4);
  float* adst  = (float*)p;    p += alignup((size_t)N * HEADS * 4);
  float* amax  = (float*)p;    p += alignup((size_t)N * HEADS * 4);
  float* den   = (float*)p;    p += alignup((size_t)N * HEADS * 4);
  float* alph  = (float*)p;    p += alignup((size_t)ETOT * HEADS * 4);
  float* agg   = (float*)p;    p += alignup((size_t)N * DOUT * 4);

  // 1. f16 conversions
  cvt_x_f16<<<(N * DIN + 255) / 256, 256, 0, stream>>>(x, xh, N * DIN);
  cvt_w_t_f16<<<(DIN * DOUT + 255) / 256, 256, 0, stream>>>(W, wt);

  // 2. xp = x @ W on WMMA (A tile via Tensor Data Mover)
  gat_gemm_xw<<<(N + 15) / 16, 128, 0, stream>>>(xh, wt, xp, N);

  // 3. per-node attention coefficients
  gat_attn_coeff<<<(N * HEADS + 255) / 256, 256, 0, stream>>>(xp, att_src, att_dst, asrc, adst, N);

  // 4. init accumulators
  gat_init<<<(N * DOUT + 255) / 256, 256, 0, stream>>>(agg, amax, den, N);

  // 5-7. segment softmax + aggregate
  gat_edge_alpha<<<(ETOT * HEADS + 255) / 256, 256, 0, stream>>>(src, dst, asrc, adst, alph, amax, E, N);
  gat_edge_exp<<<(ETOT * HEADS + 255) / 256, 256, 0, stream>>>(dst, amax, alph, den, E, N);
  gat_scatter<<<(ETOT + 7) / 8, 256, 0, stream>>>(src, dst, xp, alph, den, agg, E, N);

  // 8. bias + leaky_relu(0.01) in place -> h
  gat_bias_act<<<(N * DOUT + 255) / 256, 256, 0, stream>>>(agg, bias, N);

  // 9. edge scores
  edge_score<<<(E + 7) / 8, 256, 0, stream>>>(src, dst, agg, fc1W, fc1b, out, E);
}
